// Attention_41223096107440
// MI455X (gfx1250) — matrix operand order repaired and checked by reading, not compiled, not run on hardware
//
#include <hip/hip_runtime.h>

#define DEVINL __device__ __forceinline__

typedef __bf16 bf16;
typedef __attribute__((ext_vector_type(16))) __bf16 v16bf;
typedef __attribute__((ext_vector_type(8)))  float  v8f;
typedef __attribute__((ext_vector_type(4)))  unsigned int uint32x4;
typedef __attribute__((ext_vector_type(8)))  int int32x8;
typedef __attribute__((ext_vector_type(4)))  int int32x4;

constexpr int Bc = 2, Tc = 2048, Cc = 4096, Hc = 32, KVc = 8, Dc = 128;
constexpr int PS = 24;  // fragment lane stride (elems): 48B = 16B-aligned, 12 banks

#if defined(__has_builtin)
#if __has_builtin(__builtin_amdgcn_tensor_load_to_lds)
#define HAVE_TDM 1
#endif
#endif
#ifndef HAVE_TDM
#define HAVE_TDM 0
#endif

#if HAVE_TDM
constexpr bool kUseTdm = true;
#else
constexpr bool kUseTdm = false;
#endif

DEVINL v8f wmma_bf16(v16bf a, v16bf b, v8f c) {
  // D = A(16x32 bf16) * B(32x16 bf16) + C(16x16 f32)
  return __builtin_amdgcn_wmma_f32_16x16x32_bf16(false, a, false, b, (short)0, c,
                                                 false, false);
}
DEVINL v8f v8f_zero() {
  v8f z;
#pragma unroll
  for (int i = 0; i < 8; ++i) z[i] = 0.f;
  return z;
}
// Fragment-major LDS addressing for B operands: lane (half,l15) owns 16
// contiguous K values at ((frag*2+half)*16+l15)*PS -> 2x ds_load_b128, and
// element j maps to K = half*16 + j (same contiguous k order as the A fragments).
DEVINL int fragB(int frag, int half, int l15) { return ((frag * 2 + half) * 16 + l15) * PS; }

// --- Tensor Data Mover: 2D bf16 tile (tile_w x tile_h), row stride in elems,
// LDS padding: 4 dwords after every 16 dwords (row pitch 32 -> 40 elems). ---
DEVINL void tdm_load_tile_a(const bf16* gtile, unsigned ldsOff, int K, int M) {
#if HAVE_TDM
  const unsigned long long ga = (unsigned long long)(size_t)gtile;
  uint32x4 g0;
  g0.x = 1u;                                           // count=1 (valid), user mode
  g0.y = ldsOff;                                       // lds_addr (bytes)
  g0.z = (unsigned)ga;                                 // global_addr[31:0]
  g0.w = (unsigned)((ga >> 32) & 0x1ffffffu) | (2u << 30);  // addr[56:32] | type=2
  int32x8 g1;
  g1[0] = (1 << 16)        // data_size = 1 -> 2 bytes
        | (1 << 20)        // pad_enable
        | (3 << 22)        // pad_interval: every 16 dwords (=one 32-elem row)
        | (3 << 25);       // pad_amount: 4 dwords (=8 bf16 elems)
  g1[1] = (int)(((unsigned)K & 0xffffu) << 16);                         // tensor_dim0[15:0]
  g1[2] = (int)(((unsigned)K >> 16) & 0xffffu) | (int)(((unsigned)M & 0xffffu) << 16);
  g1[3] = (int)(((unsigned)M >> 16) & 0xffffu) | (32 << 16);            // tile_dim0 = 32
  g1[4] = 128;                                                          // tile_dim1 = 128
  g1[5] = K;                                                            // dim0_stride[31:0]
  g1[6] = 0;
  g1[7] = 0;
  int32x4 gz = {0, 0, 0, 0};
#if defined(__clang_major__) && (__clang_major__ >= 23)
  int32x8 gz8 = {0, 0, 0, 0, 0, 0, 0, 0};
  __builtin_amdgcn_tensor_load_to_lds(g0, g1, gz, gz, gz8, 0);
#else
  __builtin_amdgcn_tensor_load_to_lds(g0, g1, gz, gz, 0);
#endif
#else
  (void)gtile; (void)ldsOff; (void)K; (void)M;
#endif
}
DEVINL void tdm_wait() {
#if HAVE_TDM
#if __has_builtin(__builtin_amdgcn_s_wait_tensorcnt)
  __builtin_amdgcn_s_wait_tensorcnt(0);
#else
  asm volatile("s_wait_tensorcnt 0x0" ::: "memory");
#endif
#endif
}
DEVINL unsigned lds_off(const void* p) {  // LDS aperture: low 32 bits = byte offset
  return (unsigned)(size_t)p;
}

// ---------------------------------------------------------------------------
// Tiled GEMM: C[MxN] = A[MxK] * B[KxN], bf16 WMMA, f32 accumulation.
// Double-buffered LDS (one barrier/iter). A-side: TDM descriptor DMA (bf16 A)
// or register staging; B-side: register-staged f32 -> bf16, frag-major LDS.
// OUT_MODE: 0 = f32 row-major, 1 = bf16 packed [b][head][t][d],
//           2 = bf16 packed transposed [b][head][d][t] (for V).
// ---------------------------------------------------------------------------
template <bool A_BF16, int OUT_MODE>
__global__ __launch_bounds__(256) void gemm_wmma(const void* __restrict__ Ap,
                                                 const float* __restrict__ Bp,
                                                 void* __restrict__ Cp,
                                                 int M, int N, int K) {
  alignas(16) __shared__ bf16 sA[2][128 * 40];          // rows padded 32->40
  alignas(16) __shared__ bf16 sB[2][8 * 2 * 16 * PS];   // frag-major

  const int tid  = threadIdx.x;
  const int lane = tid & 31, wid = tid >> 5;
  const int half = lane >> 4, l15 = lane & 15;
  const int wm = wid & 3;   // 4 waves x 32 rows
  const int wn = wid >> 2;  // 2 waves x 64 cols
  const int bm = blockIdx.y * 128, bn = blockIdx.x * 128;

  v8f acc[2][4];
#pragma unroll
  for (int mi = 0; mi < 2; ++mi)
#pragma unroll
    for (int ni = 0; ni < 4; ++ni) acc[mi][ni] = v8f_zero();

  const int arow = tid >> 1, acol = (tid & 1) * 16;  // 128x32 A tile
  const int brow = tid >> 3, bcol = (tid & 7) * 16;  // 32x128 B tile
  const int nblk = tid & 7;
  const int bhalf = brow >> 4, bkin = brow & 15;

  const int nIter = K / 32;
  float4 ra[4];
  uint4  ua[2];
  float4 rb[4];

  // ---- prologue: tile 0 -> LDS[0] ----
  {
    if constexpr (A_BF16) {
      if (kUseTdm) {
        if (wid == 0) {
          tdm_load_tile_a((const bf16*)Ap + (size_t)bm * K, lds_off(&sA[0][0]), K, M);
          tdm_wait();
        }
      } else {
        const uint4* ga = (const uint4*)((const bf16*)Ap + (size_t)(bm + arow) * K + acol);
        ua[0] = ga[0];
        ua[1] = ga[1];
        uint4* la = (uint4*)(&sA[0][arow * 40 + acol]);
        la[0] = ua[0];
        la[1] = ua[1];
      }
    } else {
      const float4* ga = (const float4*)((const float*)Ap + (size_t)(bm + arow) * K + acol);
#pragma unroll
      for (int i = 0; i < 4; ++i) ra[i] = ga[i];
#pragma unroll
      for (int i = 0; i < 4; ++i) {
        float4 fv = ra[i];
        const int o = arow * 40 + acol + i * 4;
        sA[0][o + 0] = (bf16)fv.x; sA[0][o + 1] = (bf16)fv.y;
        sA[0][o + 2] = (bf16)fv.z; sA[0][o + 3] = (bf16)fv.w;
      }
    }
    const float4* gb = (const float4*)(Bp + (size_t)brow * N + bn + bcol);
#pragma unroll
    for (int i = 0; i < 4; ++i) rb[i] = gb[i];
#pragma unroll
    for (int i = 0; i < 4; ++i) {
      float4 fv = rb[i];
      const int lb = (nblk * 2 + bhalf) * 16;
      sB[0][(lb + i * 4 + 0) * PS + bkin] = (bf16)fv.x;
      sB[0][(lb + i * 4 + 1) * PS + bkin] = (bf16)fv.y;
      sB[0][(lb + i * 4 + 2) * PS + bkin] = (bf16)fv.z;
      sB[0][(lb + i * 4 + 3) * PS + bkin] = (bf16)fv.w;
    }
  }
  __syncthreads();

  for (int it = 0; it < nIter; ++it) {
    const int cur = it & 1;
    const bool hasNext = (it + 1) < nIter;

    // ---- stage tile it+1 (A: TDM or regs; B: regs), prefetch it+2 ----
    if (hasNext) {
      const int k0 = (it + 1) * 32;
      if constexpr (A_BF16) {
        if (kUseTdm) {
          if (wid == 0)
            tdm_load_tile_a((const bf16*)Ap + (size_t)bm * K + k0,
                            lds_off(&sA[cur ^ 1][0]), K, M);
        } else {
          const uint4* ga = (const uint4*)((const bf16*)Ap + (size_t)(bm + arow) * K + k0 + acol);
          ua[0] = ga[0];
          ua[1] = ga[1];
          if (it + 2 < nIter)
            __builtin_prefetch((const bf16*)Ap + (size_t)(bm + arow) * K + k0 + 32 + acol, 0, 1);
        }
      } else {
        const float4* ga = (const float4*)((const float*)Ap + (size_t)(bm + arow) * K + k0 + acol);
#pragma unroll
        for (int i = 0; i < 4; ++i) ra[i] = ga[i];
        if (it + 2 < nIter)
          __builtin_prefetch((const float*)Ap + (size_t)(bm + arow) * K + k0 + 32 + acol, 0, 1);
      }
      const float4* gb = (const float4*)(Bp + (size_t)(k0 + brow) * N + bn + bcol);
#pragma unroll
      for (int i = 0; i < 4; ++i) rb[i] = gb[i];
      if (it + 2 < nIter)
        __builtin_prefetch(Bp + (size_t)(k0 + 32 + brow) * N + bn + bcol, 0, 1);
    }

    // ---- compute from LDS[cur]: batch all fragment loads, then 8 WMMAs ----
    v16bf af[2], bfr[4];
#pragma unroll
    for (int mi = 0; mi < 2; ++mi) {
      const bf16* p = &sA[cur][(wm * 32 + mi * 16 + l15) * 40];
#pragma unroll
      for (int j = 0; j < 8; ++j) af[mi][j] = p[half * 16 + j];
#pragma unroll
      for (int j = 0; j < 8; ++j) af[mi][8 + j] = p[half * 16 + 8 + j];
    }
#pragma unroll
    for (int ni = 0; ni < 4; ++ni) {
      const bf16* p = &sB[cur][fragB(wn * 4 + ni, half, l15)];
#pragma unroll
      for (int j = 0; j < 16; ++j) bfr[ni][j] = p[j];
    }
#pragma unroll
    for (int mi = 0; mi < 2; ++mi)
#pragma unroll
      for (int ni = 0; ni < 4; ++ni)
        acc[mi][ni] = wmma_bf16(af[mi], bfr[ni], acc[mi][ni]);

    // ---- store staged regs to LDS[cur^1]; fence TDM before barrier ----
    if (hasNext) {
      const int nxt = cur ^ 1;
      if constexpr (A_BF16) {
        if (!kUseTdm) {
          uint4* la = (uint4*)(&sA[nxt][arow * 40 + acol]);
          la[0] = ua[0];
          la[1] = ua[1];
        }
      } else {
#pragma unroll
        for (int i = 0; i < 4; ++i) {
          float4 fv = ra[i];
          const int o = arow * 40 + acol + i * 4;
          sA[nxt][o + 0] = (bf16)fv.x; sA[nxt][o + 1] = (bf16)fv.y;
          sA[nxt][o + 2] = (bf16)fv.z; sA[nxt][o + 3] = (bf16)fv.w;
        }
      }
#pragma unroll
      for (int i = 0; i < 4; ++i) {
        float4 fv = rb[i];
        const int lb = (nblk * 2 + bhalf) * 16;
        sB[nxt][(lb + i * 4 + 0) * PS + bkin] = (bf16)fv.x;
        sB[nxt][(lb + i * 4 + 1) * PS + bkin] = (bf16)fv.y;
        sB[nxt][(lb + i * 4 + 2) * PS + bkin] = (bf16)fv.z;
        sB[nxt][(lb + i * 4 + 3) * PS + bkin] = (bf16)fv.w;
      }
      if constexpr (A_BF16) {
        if (kUseTdm && wid == 0) tdm_wait();
      }
    }
    __syncthreads();
  }

  // ---- epilogue ----
#pragma unroll
  for (int mi = 0; mi < 2; ++mi) {
#pragma unroll
    for (int ni = 0; ni < 4; ++ni) {
      const int row0 = bm + wm * 32 + mi * 16 + half * 8;
      const int col  = bn + wn * 64 + ni * 16 + l15;
#pragma unroll
      for (int r = 0; r < 8; ++r) {
        const int row = row0 + r;
        if constexpr (OUT_MODE == 0) {
          ((float*)Cp)[(size_t)row * N + col] = acc[mi][ni][r];
        } else {
          const int bb = row >> 11, t = row & (Tc - 1);
          const int hd = col >> 7, d = col & 127;
          const int nh = N >> 7;
          if constexpr (OUT_MODE == 1)
            ((bf16*)Cp)[(((size_t)(bb * nh + hd)) * Tc + t) * 128 + d] = (bf16)acc[mi][ni][r];
          else  // transposed [b][head][d][t] for V
            ((bf16*)Cp)[(((size_t)(bb * nh + hd)) * 128 + d) * Tc + t] = (bf16)acc[mi][ni][r];
        }
      }
    }
  }
}

// ---------------------------------------------------------------------------
// In-place RoPE on packed bf16 [b][head][t][d]; pair (d, d+64) shares angle.
// ---------------------------------------------------------------------------
__global__ void rope_inplace(bf16* buf, int npairs, float scale, float posScale) {
  const int p = blockIdx.x * blockDim.x + threadIdx.x;
  if (p >= npairs) return;
  const int d  = p & 63;
  const int t  = (p >> 6) & (Tc - 1);
  const int bh = p >> 17;  // per-(b,head) chunk = T*64 = 2^17 pairs
  const size_t base = ((size_t)bh * Tc + t) * 128;
  const float x0 = (float)buf[base + d];
  const float x1 = (float)buf[base + d + 64];
  const float invf = __expf(-(float)d * (9.210340371976184f / 64.0f));  // 10000^(-d/64)
  const float ang = (float)t * posScale * invf;
  float s, c;
  __sincosf(ang, &s, &c);
  buf[base + d]      = (bf16)((x0 * c - x1 * s) * scale);
  buf[base + d + 64] = (bf16)((x1 * c + x0 * s) * scale);
}

// ---------------------------------------------------------------------------
// Causal flash attention (GQA 4:1). 4 waves, 64 Q rows/block, 64-wide K/V
// tiles in fragment-major LDS; online softmax in f32 with 16-lane shfl_xor
// reductions. V consumed in transposed [d][t] global layout.
// ---------------------------------------------------------------------------
__global__ __launch_bounds__(128) void flash_attn(const bf16* __restrict__ qb,
                                                  const bf16* __restrict__ kb,
                                                  const bf16* __restrict__ vbT,
                                                  bf16* __restrict__ yb) {
  alignas(16) __shared__ bf16 sK[16 * 2 * 16 * PS];   // frags (sn,kk): K[s][d]
  alignas(16) __shared__ bf16 sV[16 * 2 * 16 * PS];   // frags (dt,ks): V^T[d][s]
  alignas(16) __shared__ bf16 sP[4 * 16 * 72];        // per-wave P, rows padded

  const int tid  = threadIdx.x;
  const int lane = tid & 31, wave = tid >> 5;
  const int half = lane >> 4, l15 = lane & 15;
  const int qt = blockIdx.x, h = blockIdx.y, b = blockIdx.z;
  const int kvh = h >> 2;
  const int qrow0 = qt * 64 + wave * 16;

  // Q fragments (A layout) in registers for the whole block
  v16bf aq[4];
  {
    const bf16* qp = qb + (((size_t)(b * Hc + h)) * Tc + qrow0 + l15) * Dc;
#pragma unroll
    for (int kk = 0; kk < 4; ++kk) {
#pragma unroll
      for (int j = 0; j < 8; ++j) aq[kk][j] = qp[kk * 32 + half * 16 + j];
#pragma unroll
      for (int j = 0; j < 8; ++j) aq[kk][8 + j] = qp[kk * 32 + half * 16 + 8 + j];
    }
  }

  v8f acc[8];
#pragma unroll
  for (int i = 0; i < 8; ++i) acc[i] = v8f_zero();
  float rowm[8], rowl[8];
#pragma unroll
  for (int r = 0; r < 8; ++r) { rowm[r] = -3.0e38f; rowl[r] = 0.f; }

  const bf16* kTile = kb + ((size_t)(b * KVc + kvh) * Tc) * Dc;   // [s][d]
  const bf16* vT    = vbT + ((size_t)(b * KVc + kvh) * Dc) * Tc;  // [d][t]

  for (int st = 0; st <= qt; ++st) {
    const int s0 = st * 64;
    // ---- stage K tile (64s x 128d) and V^T tile (128d x 64s), frag-major ----
#pragma unroll
    for (int j = 0; j < 4; ++j) {
      const int ci = tid + j * 128;  // 512 chunks of 16 elems each
      {  // K: s = ci>>3, 16-d chunk dc = ci&7
        const int s = ci >> 3, dc = ci & 7;
        const int kk = dc >> 1, hc = dc & 1;
        const uint4* g = (const uint4*)(kTile + (size_t)(s0 + s) * 128 + dc * 16);
        uint4* dsp = (uint4*)(&sK[fragB((s >> 4) * 4 + kk, hc, s & 15)]);
        dsp[0] = g[0];
        dsp[1] = g[1];
      }
      {  // V^T: d = ci>>2, 16-s chunk sc = ci&3
        const int d = ci >> 2, sc = ci & 3;
        const int ks = sc >> 1, hc = sc & 1;
        const uint4* g = (const uint4*)(vT + (size_t)d * Tc + s0 + sc * 16);
        uint4* dsp = (uint4*)(&sV[fragB((d >> 4) * 2 + ks, hc, d & 15)]);
        dsp[0] = g[0];
        dsp[1] = g[1];
      }
    }
    if (st < qt) {  // global_prefetch_b8 next tiles
      __builtin_prefetch((const char*)(kTile + (size_t)(s0 + 64) * 128) + tid * 128, 0, 1);
      __builtin_prefetch(vT + (size_t)tid * Tc + s0 + 64, 0, 1);
    }
    __syncthreads();

    // ---- scores: per sn, batch 4 B-frags then 4 chained WMMAs ----
    v8f sc[4];
#pragma unroll
    for (int sn = 0; sn < 4; ++sn) {
      v16bf bk[4];
#pragma unroll
      for (int kk = 0; kk < 4; ++kk) {
        const bf16* p = &sK[fragB(sn * 4 + kk, half, l15)];
#pragma unroll
        for (int j = 0; j < 16; ++j) bk[kk][j] = p[j];
      }
      v8f t = v8f_zero();
#pragma unroll
      for (int kk = 0; kk < 4; ++kk) t = wmma_bf16(aq[kk], bk[kk], t);
      const int sg = s0 + sn * 16 + l15;
#pragma unroll
      for (int r = 0; r < 8; ++r) {
        const int qg = qrow0 + half * 8 + r;
        if (sg > qg) t[r] = -3.0e38f;  // causal mask
      }
      sc[sn] = t;
    }

    // ---- online softmax ----
    float mnew[8], alpha[8];
#pragma unroll
    for (int r = 0; r < 8; ++r) {
      float mx = fmaxf(fmaxf(sc[0][r], sc[1][r]), fmaxf(sc[2][r], sc[3][r]));
#pragma unroll
      for (int off = 1; off < 16; off <<= 1) mx = fmaxf(mx, __shfl_xor(mx, off, 32));
      mnew[r]  = fmaxf(rowm[r], mx);
      alpha[r] = __expf(rowm[r] - mnew[r]);
      rowm[r]  = mnew[r];
    }
#pragma unroll
    for (int r = 0; r < 8; ++r) {
      float sum = 0.f;
#pragma unroll
      for (int sn = 0; sn < 4; ++sn) {
        float pv = __expf(sc[sn][r] - mnew[r]);
        sc[sn][r] = pv;
        sum += pv;
      }
#pragma unroll
      for (int off = 1; off < 16; off <<= 1) sum += __shfl_xor(sum, off, 32);
      rowl[r] = rowl[r] * alpha[r] + sum;
#pragma unroll
      for (int dt = 0; dt < 8; ++dt) acc[dt][r] *= alpha[r];
    }

    // ---- P: C-layout -> padded LDS rows -> A-layout fragments ----
#pragma unroll
    for (int sn = 0; sn < 4; ++sn)
#pragma unroll
      for (int r = 0; r < 8; ++r)
        sP[wave * 1152 + (half * 8 + r) * 72 + sn * 16 + l15] = (bf16)sc[sn][r];

    v16bf ap[2];
#pragma unroll
    for (int ks = 0; ks < 2; ++ks) {
      const bf16* pp = &sP[wave * 1152 + l15 * 72 + ks * 32];
#pragma unroll
      for (int j = 0; j < 8; ++j) ap[ks][j] = pp[half * 16 + j];
#pragma unroll
      for (int j = 0; j < 8; ++j) ap[ks][8 + j] = pp[half * 16 + 8 + j];
    }
    // ---- PV: per dt, batch both V-frags then 2 WMMAs ----
#pragma unroll
    for (int dt = 0; dt < 8; ++dt) {
      v16bf bv[2];
#pragma unroll
      for (int ks = 0; ks < 2; ++ks) {
        const bf16* p = &sV[fragB(dt * 2 + ks, half, l15)];
#pragma unroll
        for (int j = 0; j < 16; ++j) bv[ks][j] = p[j];
      }
#pragma unroll
      for (int ks = 0; ks < 2; ++ks) acc[dt] = wmma_bf16(ap[ks], bv[ks], acc[dt]);
    }
    __syncthreads();
  }

  // ---- epilogue: y = acc / l, bf16 row-major [b*T+t][h*128+d] ----
  float inv[8];
#pragma unroll
  for (int r = 0; r < 8; ++r) inv[r] = 1.0f / rowl[r];
  bf16* yp = yb + ((size_t)b * Tc + qrow0) * Cc + h * Dc;
#pragma unroll
  for (int dt = 0; dt < 8; ++dt)
#pragma unroll
    for (int r = 0; r < 8; ++r)
      yp[(size_t)(half * 8 + r) * Cc + dt * 16 + l15] = (bf16)(acc[dt][r] * inv[r]);
}

// ---------------------------------------------------------------------------
extern "C" void kernel_launch(void* const* d_in, const int* in_sizes, int n_in,
                              void* d_out, int out_size, void* d_ws, size_t ws_size,
                              hipStream_t stream) {
  (void)in_sizes; (void)n_in; (void)out_size; (void)ws_size;
  const float* x  = (const float*)d_in[0];
  // d_in[1] = dense triu mask -- causal mask applied analytically in-kernel
  const float* wq = (const float*)d_in[2];
  const float* wk = (const float*)d_in[3];
  const float* wv = (const float*)d_in[4];
  const float* wo = (const float*)d_in[5];
  float* out = (float*)d_out;

  char* ws = (char*)d_ws;
  bf16* qbuf = (bf16*)ws;                    // B*H*T*D  bf16 = 33,554,432 B
  bf16* kbuf = (bf16*)(ws + 33554432);       // B*KV*T*D bf16 =  8,388,608 B
  bf16* vbuf = (bf16*)(ws + 41943040);       // B*KV*D*T bf16 =  8,388,608 B (transposed)
  bf16* ybuf = (bf16*)(ws + 50331648);       // B*T*C    bf16 = 33,554,432 B

  const int M = Bc * Tc;  // 4096 rows

  // QKV projections -> packed bf16; V written transposed [b][kv][d][t]
  gemm_wmma<false, 1><<<dim3(Cc / 128, M / 128), 256, 0, stream>>>(
      x, wq, qbuf, M, Hc * Dc, Cc);
  gemm_wmma<false, 1><<<dim3((KVc * Dc) / 128, M / 128), 256, 0, stream>>>(
      x, wk, kbuf, M, KVc * Dc, Cc);
  gemm_wmma<false, 2><<<dim3((KVc * Dc) / 128, M / 128), 256, 0, stream>>>(
      x, wv, vbuf, M, KVc * Dc, Cc);

  // RoPE (reference scales positions by B); q absorbs 1/sqrt(D)
  const int qpairs = Bc * Hc * Tc * (Dc / 2);
  const int kpairs = Bc * KVc * Tc * (Dc / 2);
  rope_inplace<<<qpairs / 256, 256, 0, stream>>>(qbuf, qpairs,
                                                 0.08838834764831845f, (float)Bc);
  rope_inplace<<<kpairs / 256, 256, 0, stream>>>(kbuf, kpairs, 1.0f, (float)Bc);

  // causal flash attention -> ybuf (bf16 row-major)
  flash_attn<<<dim3(Tc / 64, Hc, Bc), 128, 0, stream>>>(qbuf, kbuf, vbuf, ybuf);

  // output projection (A-tile via Tensor Data Mover when available) -> f32
  gemm_wmma<true, 0><<<dim3(Cc / 128, M / 128), 256, 0, stream>>>(
      ybuf, wo, out, M, Cc, Cc);
}
